// MinimalAttention_76192719831325
// MI455X (gfx1250) — matrix operand order repaired and checked by reading, not compiled, not run on hardware
//
#include <hip/hip_runtime.h>

typedef __attribute__((ext_vector_type(16))) _Float16 v16h;
typedef __attribute__((ext_vector_type(8)))  float    v8f;

// async global->LDS path (gfx1250, ASYNCcnt) if toolchain exposes the builtins
#if __has_builtin(__builtin_amdgcn_global_load_async_to_lds_b128) && \
    __has_builtin(__builtin_amdgcn_s_wait_asynccnt)
#define HAVE_ASYNC_LDS 1
typedef int vi4 __attribute__((ext_vector_type(4)));
typedef __attribute__((address_space(1))) vi4 gvi4;   // global int4
typedef __attribute__((address_space(3))) vi4 lvi4;   // LDS int4
#else
#define HAVE_ASYNC_LDS 0
#endif

union F16Frag { v16h v; uint4 u[2]; };

// A-fragment: 16x32 f16, lane L: M = L&15; VGPR0-3 hold K = (L>=16?8:0)+0..7,
// VGPR4-7 hold K = 16 + (L>=16?8:0)+0..7  (ISA 7.12.2, 16-bit A-matrix)
__device__ __forceinline__ v16h ld_frag_a(const _Float16* p, int ld) {
  int lane = threadIdx.x & 31;
  int m  = lane & 15;
  int k0 = (lane >> 4) << 3;            // 0 or 8
  const _Float16* row = p + m * ld;
  F16Frag f;
  f.u[0] = *(const uint4*)(row + k0);        // K = k0 .. k0+7
  f.u[1] = *(const uint4*)(row + k0 + 16);   // K = k0+16 .. k0+23
  return f.v;
}

// B-fragment: 32x16 f16 stored as Bt[N][K]; lane L: N = L&15,
// elements 0-7 hold K = (L>=16?8:0)+0..7, elements 8-15 hold K = 16 + (L>=16?8:0)+0..7
__device__ __forceinline__ v16h ld_frag_b(const _Float16* p, int ld) {
  int lane = threadIdx.x & 31;
  int n  = lane & 15;
  int k0 = (lane >> 4) << 3;            // 0 or 8
  const _Float16* row = p + n * ld + k0;
  F16Frag f;
  f.u[0] = *(const uint4*)(row);
  f.u[1] = *(const uint4*)(row + 16);
  return f.v;
}

__device__ __forceinline__ v8f wmma_f16(v16h a, v16h b, v8f c) {
  return __builtin_amdgcn_wmma_f32_16x16x32_f16(false, a, false, b, (short)0, c, false, false);
}

// DPP16 xor-butterfly stage (pure VALU, no DS round-trip)
template <int CTRL>
__device__ __forceinline__ float dpp_f32(float x) {
  return __int_as_float(__builtin_amdgcn_update_dpp(0, __float_as_int(x), CTRL, 0xF, 0xF, true));
}
// max over the 16-lane half-group: xor1 (quad_perm[1,0,3,2]=0xB1),
// xor2 (quad_perm[2,3,0,1]=0x4E), xor7 (row_half_mirror=0x141), xor15 (row_mirror=0x140)
__device__ __forceinline__ float rowmax16(float x) {
  x = fmaxf(x, dpp_f32<0xB1>(x));
  x = fmaxf(x, dpp_f32<0x4E>(x));
  x = fmaxf(x, dpp_f32<0x141>(x));
  x = fmaxf(x, dpp_f32<0x140>(x));
  return x;
}

// ---------------- convert helpers ----------------
// 8 floats -> 8 f16 per thread (b128 store)
__global__ void cvt_f16_vec_kernel(const float* __restrict__ in, _Float16* __restrict__ out, int n8) {
  int i = blockIdx.x * 256 + threadIdx.x;
  if (i < n8) {
    const float4* p = (const float4*)in + (size_t)i * 2;
    float4 a = p[0], b = p[1];
    union { uint4 u; _Float16 h[8]; } o;
    o.h[0] = (_Float16)a.x; o.h[1] = (_Float16)a.y; o.h[2] = (_Float16)a.z; o.h[3] = (_Float16)a.w;
    o.h[4] = (_Float16)b.x; o.h[5] = (_Float16)b.y; o.h[6] = (_Float16)b.z; o.h[7] = (_Float16)b.w;
    *((uint4*)out + i) = o.u;
  }
}

// in[R][C] f32 -> out[C][R] f16
__global__ void transpose_cvt_kernel(const float* __restrict__ in, _Float16* __restrict__ out,
                                     int R, int C) {
  int i = blockIdx.x * 256 + threadIdx.x;
  if (i < R * C) {
    int r = i / C;
    int c = i - r * C;
    out[(size_t)c * R + r] = (_Float16)in[i];
  }
}

// ---------------- tiled WMMA GEMM: C[M,N] = A[M,K] * Bt[N,K]^T + bias ----------------
// MODE 0: write f32 Cout.  MODE 1: QKV epilogue -> split f16 Q/K/V in [B*H][S][64],
// with 0.125*log2(e) folded into Q (softmax done in exp2 domain).
template <int MODE>
__global__ __launch_bounds__(256)
void gemm128_wmma(const _Float16* __restrict__ A, const _Float16* __restrict__ Bt,
                  const float* __restrict__ bias, float* __restrict__ Cout,
                  _Float16* __restrict__ Qo, _Float16* __restrict__ Ko, _Float16* __restrict__ Vo,
                  int M, int N, int K) {
  const int bm  = blockIdx.y * 128;
  const int bn  = blockIdx.x * 128;
  const int tid = threadIdx.x;
  const int w   = tid >> 5;
  const int wm  = (w >> 2) * 64;   // 0 / 64
  const int wn  = (w & 3) * 32;    // 0,32,64,96

  const int c0   = tid * 2;
  const int row0 = c0 >> 2,       col0 = (c0 & 3) * 8;
  const int row1 = (c0 + 1) >> 2, col1 = ((c0 + 1) & 3) * 8;

  v8f acc[4][2] = {};

#if HAVE_ASYNC_LDS
  // double-buffered LDS, tiles streamed with GLOBAL_LOAD_ASYNC_TO_LDS_B128 (ASYNCcnt)
  __shared__ alignas(16) _Float16 As[2][128 * 48];
  __shared__ alignas(16) _Float16 Bs[2][128 * 48];

#define AISSUE(kb, bf)                                                                            \
  {                                                                                               \
    __builtin_amdgcn_global_load_async_to_lds_b128(                                               \
        (gvi4*)&A [(size_t)(bm + row0) * K + (kb) + col0], (lvi4*)&As[bf][row0 * 48 + col0], 0, 0); \
    __builtin_amdgcn_global_load_async_to_lds_b128(                                               \
        (gvi4*)&A [(size_t)(bm + row1) * K + (kb) + col1], (lvi4*)&As[bf][row1 * 48 + col1], 0, 0); \
    __builtin_amdgcn_global_load_async_to_lds_b128(                                               \
        (gvi4*)&Bt[(size_t)(bn + row0) * K + (kb) + col0], (lvi4*)&Bs[bf][row0 * 48 + col0], 0, 0); \
    __builtin_amdgcn_global_load_async_to_lds_b128(                                               \
        (gvi4*)&Bt[(size_t)(bn + row1) * K + (kb) + col1], (lvi4*)&Bs[bf][row1 * 48 + col1], 0, 0); \
  }

  AISSUE(0, 0);
  int ib = 0;
  for (int kb = 0; kb < K; kb += 32, ib ^= 1) {
    __syncthreads();                          // all waves done reading buffer ib^1
    if (kb + 32 < K) {
      AISSUE(kb + 32, ib ^ 1);
      __builtin_amdgcn_s_wait_asynccnt(4);    // own tile-ib loads complete (FIFO)
    } else {
      __builtin_amdgcn_s_wait_asynccnt(0);
    }
    __syncthreads();                          // everyone's tile-ib data visible in LDS

    v16h af[4], bf[2];
#pragma unroll
    for (int ms = 0; ms < 4; ++ms) af[ms] = ld_frag_a(&As[ib][(wm + ms * 16) * 48], 48);
#pragma unroll
    for (int ns = 0; ns < 2; ++ns) bf[ns] = ld_frag_b(&Bs[ib][(wn + ns * 16) * 48], 48);
#pragma unroll
    for (int ms = 0; ms < 4; ++ms)
#pragma unroll
      for (int ns = 0; ns < 2; ++ns)
        acc[ms][ns] = wmma_f16(af[ms], bf[ns], acc[ms][ns]);
  }
#undef AISSUE
#else
  // fallback: register-staged software pipeline
  __shared__ alignas(16) _Float16 As[128 * 48];
  __shared__ alignas(16) _Float16 Bs[128 * 48];
  uint4 ra[2], rb[2];

#define GFETCH(kb)                                                             \
  {                                                                            \
    ra[0] = *(const uint4*)&A [(size_t)(bm + row0) * K + (kb) + col0];         \
    ra[1] = *(const uint4*)&A [(size_t)(bm + row1) * K + (kb) + col1];         \
    rb[0] = *(const uint4*)&Bt[(size_t)(bn + row0) * K + (kb) + col0];         \
    rb[1] = *(const uint4*)&Bt[(size_t)(bn + row1) * K + (kb) + col1];         \
  }

  GFETCH(0);
  for (int kb = 0; kb < K; kb += 32) {
    *(uint4*)&As[row0 * 48 + col0] = ra[0];
    *(uint4*)&As[row1 * 48 + col1] = ra[1];
    *(uint4*)&Bs[row0 * 48 + col0] = rb[0];
    *(uint4*)&Bs[row1 * 48 + col1] = rb[1];
    __syncthreads();
    if (kb + 32 < K) GFETCH(kb + 32);

    v16h af[4], bf[2];
#pragma unroll
    for (int ms = 0; ms < 4; ++ms) af[ms] = ld_frag_a(&As[(wm + ms * 16) * 48], 48);
#pragma unroll
    for (int ns = 0; ns < 2; ++ns) bf[ns] = ld_frag_b(&Bs[(wn + ns * 16) * 48], 48);
#pragma unroll
    for (int ms = 0; ms < 4; ++ms)
#pragma unroll
      for (int ns = 0; ns < 2; ++ns)
        acc[ms][ns] = wmma_f16(af[ms], bf[ns], acc[ms][ns]);
    __syncthreads();
  }
#undef GFETCH
#endif

  const int lane = tid & 31;
  const int ncol = lane & 15;
  const int mref = (lane >> 4) * 8;
#pragma unroll
  for (int ms = 0; ms < 4; ++ms) {
#pragma unroll
    for (int ns = 0; ns < 2; ++ns) {
#pragma unroll
      for (int v = 0; v < 8; ++v) {
        int   row  = bm + wm + ms * 16 + mref + v;
        int   coln = bn + wn + ns * 16 + ncol;
        float val  = acc[ms][ns][v] + bias[coln];
        if (MODE == 0) {
          Cout[(size_t)row * N + coln] = val;
        } else {
          int t = coln >> 10;          // 0=Q 1=K 2=V
          int h = (coln >> 6) & 15;
          int d = coln & 63;
          int b = row >> 11;
          int s = row & 2047;
          if (t == 0) val *= 0.18033688011112042f;  // (1/sqrt(64)) * log2(e)
          _Float16* dst = (t == 0) ? Qo : (t == 1 ? Ko : Vo);
          dst[((size_t)(b * 16 + h) * 2048 + s) * 64 + d] = (_Float16)val;
        }
      }
    }
  }
}

// ---------------- flash attention: one (b,h) per blockIdx.y, 128 q-rows per block ----------------
// Softmax denominator computed as a 5th WMMA output tile (P @ ones) -> no sum butterfly,
// row-max reduced with DPP16 butterflies (VALU only, co-executes with WMMA).
__global__ __launch_bounds__(256)
void flash_attn_wmma(const _Float16* __restrict__ Qb, const _Float16* __restrict__ Kb,
                     const _Float16* __restrict__ Vb, _Float16* __restrict__ AO) {
  __shared__ alignas(16) _Float16 Qs[128 * 80];
  __shared__ alignas(16) _Float16 Ks[64 * 80];
  __shared__ alignas(16) _Float16 Vt[64 * 80];   // transposed: [d][key]
  __shared__ alignas(16) _Float16 Ps[128 * 80];  // per-wave 16-row P staging

  const int    bh    = blockIdx.y;           // b*16 + h
  const int    qbase = blockIdx.x * 128;
  const int    tid   = threadIdx.x;
  const int    w     = tid >> 5;
  const int    lane  = tid & 31;
  const size_t headOff = (size_t)bh * 2048 * 64;

  // Q tile 128x64 (4 x b128 per thread)
#pragma unroll
  for (int i = 0; i < 4; ++i) {
    int c   = tid * 4 + i;
    int row = c >> 3;
    int col = (c & 7) * 8;
    *(uint4*)&Qs[row * 80 + col] = *(const uint4*)&Qb[headOff + (size_t)(qbase + row) * 64 + col];
  }

  // constant all-ones B fragment: l-tile = P @ ones -> row sums in every column
  v16h ones;
#pragma unroll
  for (int i = 0; i < 16; ++i) ones[i] = (_Float16)1.0f;

  v8f   o[5] = {};           // o[0..3] = output d-tiles, o[4] = softmax denominator tile
  float mrun[8];
#pragma unroll
  for (int v = 0; v < 8; ++v) mrun[v] = -1e30f;

  // K/V prefetch registers (tile t+1 fetched during compute of tile t)
  const int krow = (tid * 2) >> 3,       kcol = ((tid * 2) & 7) * 8;
  const int krow1 = (tid * 2 + 1) >> 3,  kcol1 = ((tid * 2 + 1) & 7) * 8;
  const int vkey = tid >> 2, vdbase = (tid & 3) * 16;
  uint4 rk0, rk1;
  union { uint4 u[2]; _Float16 h[16]; } rv;

#define KVFETCH(kb)                                                                         \
  {                                                                                         \
    rk0 = *(const uint4*)&Kb[headOff + (size_t)((kb) + krow)  * 64 + kcol];                 \
    rk1 = *(const uint4*)&Kb[headOff + (size_t)((kb) + krow1) * 64 + kcol1];                \
    const _Float16* vs = &Vb[headOff + (size_t)((kb) + vkey) * 64 + vdbase];                \
    rv.u[0] = *(const uint4*)(vs);                                                          \
    rv.u[1] = *(const uint4*)(vs + 8);                                                      \
  }

  KVFETCH(0);
  for (int kb = 0; kb < 2048; kb += 64) {
    // stage K row-major, V transposed into LDS
    *(uint4*)&Ks[krow * 80 + kcol]   = rk0;
    *(uint4*)&Ks[krow1 * 80 + kcol1] = rk1;
#pragma unroll
    for (int i = 0; i < 16; ++i) Vt[(vdbase + i) * 80 + vkey] = rv.h[i];
    __syncthreads();
    if (kb + 64 < 2048) KVFETCH(kb + 64);

    // scores (already in log2 domain; scale folded into Q)
    v8f  sc[4] = {};
    v16h qa[2];
#pragma unroll
    for (int kk = 0; kk < 2; ++kk) qa[kk] = ld_frag_a(&Qs[(w * 16) * 80 + kk * 32], 80);
#pragma unroll
    for (int n = 0; n < 4; ++n)
#pragma unroll
      for (int kk = 0; kk < 2; ++kk)
        sc[n] = wmma_f16(qa[kk], ld_frag_b(&Ks[(n * 16) * 80 + kk * 32], 80), sc[n]);

    // online softmax: DPP row-max, exp2, rescale all 5 accumulator tiles
#pragma unroll
    for (int v = 0; v < 8; ++v) {
      float mx = fmaxf(fmaxf(sc[0][v], sc[1][v]), fmaxf(sc[2][v], sc[3][v]));
      mx = rowmax16(mx);
      float mnew = fmaxf(mrun[v], mx);
      float corr = exp2f(mrun[v] - mnew);
      mrun[v]    = mnew;
#pragma unroll
      for (int n = 0; n < 4; ++n) sc[n][v] = exp2f(sc[n][v] - mnew);
#pragma unroll
      for (int t = 0; t < 5; ++t) o[t][v] *= corr;
    }

    // stage P (f16) into this wave's LDS strip for A-fragment reload
    {
      int mref = (lane >> 4) * 8;
      int cl   = lane & 15;
#pragma unroll
      for (int n = 0; n < 4; ++n)
#pragma unroll
        for (int v = 0; v < 8; ++v)
          Ps[(w * 16 + mref + v) * 80 + n * 16 + cl] = (_Float16)sc[n][v];
    }

    // O += P @ V ; l += P @ ones
    v16h pa[2];
#pragma unroll
    for (int kk = 0; kk < 2; ++kk) pa[kk] = ld_frag_a(&Ps[(w * 16) * 80 + kk * 32], 80);
#pragma unroll
    for (int dn = 0; dn < 4; ++dn)
#pragma unroll
      for (int kk = 0; kk < 2; ++kk)
        o[dn] = wmma_f16(pa[kk], ld_frag_b(&Vt[(dn * 16) * 80 + kk * 32], 80), o[dn]);
#pragma unroll
    for (int kk = 0; kk < 2; ++kk)
      o[4] = wmma_f16(pa[kk], ones, o[4]);
    __syncthreads();
  }
#undef KVFETCH

  // epilogue: normalize by the WMMA-computed denominator, write f16 AO in [B,S,H*HD]
  const int b    = bh >> 4;
  const int h    = bh & 15;
  const int mref = (lane >> 4) * 8;
  const int cl   = lane & 15;
#pragma unroll
  for (int v = 0; v < 8; ++v) {
    float inv = 1.0f / o[4][v];
    int   s   = qbase + w * 16 + mref + v;
#pragma unroll
    for (int dn = 0; dn < 4; ++dn) {
      float val = o[dn][v] * inv;
      AO[((size_t)(b * 2048 + s) * 16 + h) * 64 + dn * 16 + cl] = (_Float16)val;
    }
  }
}

// ---------------- host launcher ----------------
extern "C" void kernel_launch(void* const* d_in, const int* in_sizes, int n_in,
                              void* d_out, int out_size, void* d_ws, size_t ws_size,
                              hipStream_t stream) {
  (void)in_sizes; (void)n_in; (void)out_size; (void)ws_size;
  const float* x      = (const float*)d_in[0];
  const float* w_qkv  = (const float*)d_in[1];
  const float* b_qkv  = (const float*)d_in[2];
  const float* w_proj = (const float*)d_in[3];
  const float* b_proj = (const float*)d_in[4];
  float*       out    = (float*)d_out;

  // f16 workspace layout (48 MB total)
  _Float16* ws     = (_Float16*)d_ws;
  _Float16* xh     = ws;                                  // 4096*1024
  _Float16* wqkvT  = xh     + (size_t)4096 * 1024;        // 3072*1024
  _Float16* wprojT = wqkvT  + (size_t)3072 * 1024;        // 1024*1024
  _Float16* Qb     = wprojT + (size_t)1024 * 1024;        // 32*2048*64
  _Float16* Kb     = Qb     + (size_t)32 * 2048 * 64;
  _Float16* Vb     = Kb     + (size_t)32 * 2048 * 64;
  _Float16* AO     = Vb     + (size_t)32 * 2048 * 64;     // 4096*1024

  cvt_f16_vec_kernel<<<2048, 256, 0, stream>>>(x, xh, (4096 * 1024) / 8);
  transpose_cvt_kernel<<<12288, 256, 0, stream>>>(w_qkv, wqkvT, 1024, 3072);
  transpose_cvt_kernel<<<4096, 256, 0, stream>>>(w_proj, wprojT, 1024, 1024);

  // QKV: [4096,1024] x [1024,3072]
  gemm128_wmma<1><<<dim3(24, 32), 256, 0, stream>>>(xh, wqkvT, b_qkv, nullptr,
                                                    Qb, Kb, Vb, 4096, 3072, 1024);
  // attention: 16 q-tiles x 32 (b,h)
  flash_attn_wmma<<<dim3(16, 32), 256, 0, stream>>>(Qb, Kb, Vb, AO);
  // projection: [4096,1024] x [1024,1024] -> f32 out
  gemm128_wmma<0><<<dim3(8, 32), 256, 0, stream>>>(AO, wprojT, b_proj, out,
                                                   nullptr, nullptr, nullptr, 4096, 1024, 1024);
}
